// ChunkTransformerLayer_9345848836380
// MI455X (gfx1250) — matrix operand order repaired and checked by reading, not compiled, not run on hardware
//
#include <hip/hip_runtime.h>
#include <math.h>

#define DEVI __device__ __forceinline__

namespace {

constexpr int Bd = 2, Td = 1024, Cd = 1024, Hd = 16, DHd = 64, LCd = 256, DFFd = 4096;
constexpr int NT  = Bd * Td;    // 2048 token rows
constexpr int NCC = Bd * LCd;   // 512 context rows
constexpr float SCALE = 0.125f; // 1/sqrt(dh=64)

typedef __attribute__((ext_vector_type(16))) __bf16          v16bf;
typedef __attribute__((ext_vector_type(8)))  float           v8f;
typedef __attribute__((ext_vector_type(8)))  unsigned short  u16x8;
typedef __attribute__((ext_vector_type(4)))  unsigned        u32x4;
typedef __attribute__((ext_vector_type(8)))  unsigned        u32x8;

union Frag { v16bf bf; u16x8 h[2]; };

DEVI unsigned short f2bf(float f) {
  union { float f; unsigned u; } v; v.f = f;
  unsigned r = v.u + 0x7fffu + ((v.u >> 16) & 1u);   // round-to-nearest-even
  return (unsigned short)(r >> 16);
}
DEVI float bf2f(unsigned short h) {
  union { unsigned u; float f; } v; v.u = ((unsigned)h) << 16;
  return v.f;
}
// A-frag pattern: 16 contiguous elements from p0 (p1 unused)
DEVI v16bf frag2(const unsigned short* p0, const unsigned short* p1) {
  (void)p1; Frag f; f.h[0] = *(const u16x8*)p0; f.h[1] = *(const u16x8*)(p0 + 8); return f.bf;
}
// B-frag pattern: 16 contiguous elements (32B)
DEVI v16bf frag16(const unsigned short* p) {
  Frag f; f.h[0] = *(const u16x8*)p; f.h[1] = *(const u16x8*)(p + 8); return f.bf;
}
DEVI v8f wmma_bf16(v16bf a, v16bf b, v8f c) {
  return __builtin_amdgcn_wmma_f32_16x16x32_bf16(false, a, false, b, (short)0, c, false, false);
}
DEVI v8f zero8() {
  v8f z;
  #pragma unroll
  for (int i = 0; i < 8; ++i) z[i] = 0.f;
  return z;
}
// LDS byte offset of a __shared__ object (flat LDS addresses: low 32 bits)
DEVI unsigned lds_off(const void* p) {
  return (unsigned)(unsigned long long)p;
}

// ---------------------------------------------------------------------------
// fp32 -> bf16 conversion (weights, context)
// ---------------------------------------------------------------------------
__global__ void k_f32_to_bf16(const float* __restrict__ src,
                              unsigned short* __restrict__ dst, int n) {
  int i = blockIdx.x * 256 + threadIdx.x;
  if (i < n) dst[i] = f2bf(src[i]);
}

// ---------------------------------------------------------------------------
// LayerNorm (eps 1e-6, no affine) + either cond-affine (g,b) or AdaLN mod
// MODE 0: out = ln(x)*g + b        MODE 1: out = ln(x)*(1+scale) + shift
// One row (C=1024) per 256-thread block.
// ---------------------------------------------------------------------------
template <int MODE>
__global__ __launch_bounds__(256)
void k_layernorm(const float* __restrict__ x,
                 const float* __restrict__ gw, const float* __restrict__ bw,
                 const unsigned short* __restrict__ ada, int shift_off, int scale_off,
                 unsigned short* __restrict__ out) {
  int row = blockIdx.x;
  const float* xr = x + (long)row * Cd;
  int t = threadIdx.x;
  float v[4]; float s = 0.f, ss = 0.f;
  #pragma unroll
  for (int i = 0; i < 4; ++i) { v[i] = xr[t + 256 * i]; s += v[i]; ss += v[i] * v[i]; }
  #pragma unroll
  for (int off = 16; off > 0; off >>= 1) { s += __shfl_down(s, off, 32); ss += __shfl_down(ss, off, 32); }
  __shared__ float smem[16];
  int wave = t >> 5, lane = t & 31;
  if (lane == 0) { smem[wave] = s; smem[8 + wave] = ss; }
  __syncthreads();
  if (t == 0) {
    float a = 0.f, b2 = 0.f;
    #pragma unroll
    for (int w = 0; w < 8; ++w) { a += smem[w]; b2 += smem[8 + w]; }
    float mean = a * (1.f / Cd);
    float var  = b2 * (1.f / Cd) - mean * mean;
    smem[0] = mean; smem[1] = rsqrtf(var + 1e-6f);
  }
  __syncthreads();
  float mean = smem[0], rstd = smem[1];
  unsigned short* orow = out + (long)row * Cd;
  #pragma unroll
  for (int i = 0; i < 4; ++i) {
    int col = t + 256 * i;
    float y = (v[i] - mean) * rstd;
    if (MODE == 0) {
      y = y * gw[col] + bw[col];
    } else {
      float sc = bf2f(ada[(long)row * 6 * Cd + scale_off + col]);
      float sh = bf2f(ada[(long)row * 6 * Cd + shift_off + col]);
      y = y * (1.f + sc) + sh;
    }
    orow[col] = f2bf(y);
  }
}

// out = x + gate * y   (gate from AdaLN buffer, bf16)
__global__ void k_resid_gate(const float* __restrict__ x, const float* __restrict__ y,
                             const unsigned short* __restrict__ ada, int gate_off,
                             float* __restrict__ out, int n) {
  int i = blockIdx.x * 256 + threadIdx.x;
  if (i >= n) return;
  int row = i / Cd, col = i - row * Cd;
  float g = bf2f(ada[(long)row * 6 * Cd + gate_off + col]);
  out[i] = x[i] + g * y[i];
}

// ---------------------------------------------------------------------------
// bf16 GEMM:  out[M,N] = act(A[M,K] @ W[K,N] + bias)
// 64x128 block tile, 128 threads (4 waves); wave w owns cols [w*32, w*32+32).
// A tile (64x32, padded rows of 40 elems) is staged by the Tensor Data Mover:
// wave 0 issues tensor_load_to_lds with a 2-D D# (pad 4 DWORDs / 16 DWORDs so
// the LDS layout matches As[64][40]), waits TENSORcnt, then the barrier
// publishes the tile to all waves.
// EPI: 0 f32+bias   1 bf16+bias   2 bf16 SiLU   3 bf16 GELU(tanh)
// ---------------------------------------------------------------------------
template <int EPI>
__global__ __launch_bounds__(128)
void k_gemm(const unsigned short* __restrict__ A, const unsigned short* __restrict__ W,
            const float* __restrict__ bias, void* __restrict__ outp,
            int M, int N, int K) {
  __shared__ unsigned short As[64][40];    // [m][k]  80B rows (16B aligned)
  __shared__ unsigned short Bs[128][40];   // [n][k]  transposed weight tile
  const int tid = threadIdx.x;
  const int wave = tid >> 5, lane = tid & 31;
  const int lrow = lane & 15, half = lane >> 4;
  const int n0 = blockIdx.x * 128, m0 = blockIdx.y * 64;

  // TDM descriptor group 1 (uniform across the whole loop)
  // data_size=2B(1<<16) | pad_enable(1<<20) | pad_interval=3:16dw(3<<22)
  // | pad_amount=3:4dw(3<<25); tensor_dim0=K, tensor_dim1=64, tile=32x64,
  // tensor_dim0_stride=K
  u32x8 g1;
  g1[0] = (1u << 16) | (1u << 20) | (3u << 22) | (3u << 25);
  g1[1] = ((unsigned)K & 0xffffu) << 16;
  g1[2] = (((unsigned)K >> 16) & 0xffffu) | (64u << 16);
  g1[3] = (32u << 16);
  g1[4] = 64u;
  g1[5] = (unsigned)K;
  g1[6] = 0u;
  g1[7] = 0u;
  const unsigned ldsA = lds_off(&As[0][0]);
  const unsigned long long abase =
      (unsigned long long)A + ((unsigned long long)m0 * (unsigned)K) * 2ull;

  v8f acc[4][2];
  #pragma unroll
  for (int m = 0; m < 4; ++m) { acc[m][0] = zero8(); acc[m][1] = zero8(); }

  for (int k0 = 0; k0 < K; k0 += 32) {
    if (wave == 0) {   // TDM: A tile 64x32 -> LDS (padded to 40-elem rows)
      unsigned long long ga = abase + (unsigned long long)k0 * 2ull;
      u32x4 g0;
      g0[0] = 1u;                               // count=1
      g0[1] = ldsA;                             // lds_addr
      g0[2] = (unsigned)(ga & 0xffffffffull);   // global_addr lo
      g0[3] = (unsigned)(ga >> 32) | (2u << 30);// global_addr hi | type=2
      asm volatile("tensor_load_to_lds %0, %1" :: "s"(g0), "s"(g1) : "memory");
    }
    { // stage W tile 32x128 transposed -> Bs[n][k] (coalesced across n)
      int n = tid;
      #pragma unroll
      for (int i = 0; i < 32; ++i)
        Bs[n][i] = W[(long)(k0 + i) * N + n0 + n];
      if (k0 + 32 < K)
        __builtin_prefetch(&W[(long)(k0 + 32) * N + n0 + n], 0, 1);
    }
    if (wave == 0) __builtin_amdgcn_s_wait_tensorcnt(0);
    __syncthreads();

    const int ab = half ? 16 : 0;
    v16bf bf0 = frag16(&Bs[wave * 32 + lrow][half ? 16 : 0]);
    v16bf bf1 = frag16(&Bs[wave * 32 + 16 + lrow][half ? 16 : 0]);
    v16bf a0 = frag2(&As[ 0 + lrow][ab], &As[ 0 + lrow][ab + 16]);
    v16bf a1 = frag2(&As[16 + lrow][ab], &As[16 + lrow][ab + 16]);
    v16bf a2 = frag2(&As[32 + lrow][ab], &As[32 + lrow][ab + 16]);
    v16bf a3 = frag2(&As[48 + lrow][ab], &As[48 + lrow][ab + 16]);
    acc[0][0] = wmma_bf16(a0, bf0, acc[0][0]);
    acc[1][0] = wmma_bf16(a1, bf0, acc[1][0]);
    acc[2][0] = wmma_bf16(a2, bf0, acc[2][0]);
    acc[3][0] = wmma_bf16(a3, bf0, acc[3][0]);
    acc[0][1] = wmma_bf16(a0, bf1, acc[0][1]);
    acc[1][1] = wmma_bf16(a1, bf1, acc[1][1]);
    acc[2][1] = wmma_bf16(a2, bf1, acc[2][1]);
    acc[3][1] = wmma_bf16(a3, bf1, acc[3][1]);
    __syncthreads();
  }

  #pragma unroll
  for (int j = 0; j < 2; ++j) {
    int col = n0 + wave * 32 + j * 16 + lrow;
    float bv = bias ? bias[col] : 0.f;
    #pragma unroll
    for (int m = 0; m < 4; ++m) {
      #pragma unroll
      for (int i = 0; i < 8; ++i) {
        int rg = m0 + m * 16 + half * 8 + i;
        float v = acc[m][j][i] + bv;
        if (EPI == 2) v = v / (1.f + __expf(-v));
        if (EPI == 3) { float c = v;
          v = 0.5f * c * (1.f + tanhf(0.7978845608f * (c + 0.044715f * c * c * c))); }
        if (EPI == 0) ((float*)outp)[(long)rg * N + col] = v;
        else ((unsigned short*)outp)[(long)rg * N + col] = f2bf(v);
      }
    }
  }
}

// ---------------------------------------------------------------------------
// Cross attention: Q[B*T,C] x KV[B*LC,2C] -> Out[B*T,C] (no mask).
// Block = 128 threads = 4 waves; each wave owns a 16-row q tile (64 rows/blk).
// ---------------------------------------------------------------------------
__global__ __launch_bounds__(128)
void k_cross_attn(const unsigned short* __restrict__ Q,
                  const unsigned short* __restrict__ KV,
                  unsigned short* __restrict__ Out) {
  __shared__ unsigned short Kt[32][72];       // [s][d]
  __shared__ unsigned short Vt[64][40];       // [d][s] transposed
  __shared__ unsigned short Ps[4][16][40];    // per-wave P tile
  const int tid = threadIdx.x, wave = tid >> 5, lane = tid & 31;
  const int lrow = lane & 15, half = lane >> 4, ab = half ? 16 : 0;
  const int b = blockIdx.x / Hd, h = blockIdx.x % Hd;
  const int t0 = blockIdx.y * 64 + wave * 16;

  const unsigned short* qrow = Q + (long)(b * Td + t0 + lrow) * Cd + h * DHd;
  v16bf aq0 = frag2(qrow + ab,      qrow + ab + 16);
  v16bf aq1 = frag2(qrow + 32 + ab, qrow + 32 + ab + 16);

  v8f o[4];
  #pragma unroll
  for (int n = 0; n < 4; ++n) o[n] = zero8();
  float mr[8], lr[8];
  #pragma unroll
  for (int i = 0; i < 8; ++i) { mr[i] = -3.0e38f; lr[i] = 0.f; }

  for (int s0 = 0; s0 < LCd; s0 += 32) {
    { // stage K (32x64) and V transposed (64x32)
      int s = tid >> 2, db = (tid & 3) * 16;
      const unsigned short* kg = KV + (long)(b * LCd + s0 + s) * 2 * Cd + h * DHd + db;
      *(u16x8*)&Kt[s][db]     = *(const u16x8*)kg;
      *(u16x8*)&Kt[s][db + 8] = *(const u16x8*)(kg + 8);
      const unsigned short* vg = kg + Cd;
      u16x8 v0 = *(const u16x8*)vg, v1 = *(const u16x8*)(vg + 8);
      #pragma unroll
      for (int i = 0; i < 8; ++i) { Vt[db + i][s] = v0[i]; Vt[db + 8 + i][s] = v1[i]; }
    }
    __syncthreads();

    v8f sc[2]; sc[0] = zero8(); sc[1] = zero8();
    #pragma unroll
    for (int c = 0; c < 2; ++c) {
      v16bf b0 = frag16(&Kt[c * 16 + lrow][half ? 16 : 0]);
      v16bf b1 = frag16(&Kt[c * 16 + lrow][32 + (half ? 16 : 0)]);
      sc[c] = wmma_bf16(aq0, b0, sc[c]);
      sc[c] = wmma_bf16(aq1, b1, sc[c]);
    }
    float mn[8];
    #pragma unroll
    for (int i = 0; i < 8; ++i) {
      sc[0][i] *= SCALE; sc[1][i] *= SCALE;
      mn[i] = fmaxf(mr[i], fmaxf(sc[0][i], sc[1][i]));
      #pragma unroll
      for (int off = 1; off < 16; off <<= 1) mn[i] = fmaxf(mn[i], __shfl_xor(mn[i], off, 32));
    }
    #pragma unroll
    for (int i = 0; i < 8; ++i) {
      float corr = (mn[i] <= -1e30f) ? 1.f : __expf(mr[i] - mn[i]);
      float p0 = __expf(sc[0][i] - mn[i]);
      float p1 = __expf(sc[1][i] - mn[i]);
      float ps = p0 + p1;
      #pragma unroll
      for (int off = 1; off < 16; off <<= 1) ps += __shfl_xor(ps, off, 32);
      lr[i] = lr[i] * corr + ps;
      mr[i] = mn[i];
      int r = half * 8 + i;
      Ps[wave][r][lrow]      = f2bf(p0);
      Ps[wave][r][16 + lrow] = f2bf(p1);
      #pragma unroll
      for (int n = 0; n < 4; ++n) o[n][i] *= corr;
    }
    v16bf ap = frag2(&Ps[wave][lrow][ab], &Ps[wave][lrow][ab + 16]);
    #pragma unroll
    for (int n = 0; n < 4; ++n) {
      v16bf bv = frag16(&Vt[n * 16 + lrow][half ? 16 : 0]);
      o[n] = wmma_bf16(ap, bv, o[n]);
    }
    __syncthreads();
  }
  #pragma unroll
  for (int n = 0; n < 4; ++n) {
    #pragma unroll
    for (int i = 0; i < 8; ++i) {
      int r = t0 + half * 8 + i;
      Out[(long)(b * Td + r) * Cd + h * DHd + n * 16 + lrow] = f2bf(o[n][i] / lr[i]);
    }
  }
}

// ---------------------------------------------------------------------------
// Star-stream self attention: causal + dependency mask.
// QKV layout [B*T, 3C] = [q | k | v], head h cols h*64..h*64+63 in each third.
// ---------------------------------------------------------------------------
__global__ __launch_bounds__(128)
void k_attn_star(const unsigned short* __restrict__ QKV,
                 const unsigned char* __restrict__ dep,
                 unsigned short* __restrict__ Out) {
  __shared__ unsigned short Kt[32][72];
  __shared__ unsigned short Vt[64][40];
  __shared__ unsigned short Ps[4][16][40];
  const int tid = threadIdx.x, wave = tid >> 5, lane = tid & 31;
  const int lrow = lane & 15, half = lane >> 4, ab = half ? 16 : 0;
  const int b = blockIdx.x / Hd, h = blockIdx.x % Hd;
  const int t0 = blockIdx.y * 64 + wave * 16;
  const float NINF = -3.0e38f;

  const unsigned short* qrow = QKV + (long)(b * Td + t0 + lrow) * 3 * Cd + h * DHd;
  v16bf aq0 = frag2(qrow + ab,      qrow + ab + 16);
  v16bf aq1 = frag2(qrow + 32 + ab, qrow + 32 + ab + 16);

  v8f o[4];
  #pragma unroll
  for (int n = 0; n < 4; ++n) o[n] = zero8();
  float mr[8], lr[8];
  #pragma unroll
  for (int i = 0; i < 8; ++i) { mr[i] = NINF; lr[i] = 0.f; }

  const int send = blockIdx.y * 64 + 64;   // causal clip: s <= max t in this block
  for (int s0 = 0; s0 < send; s0 += 32) {
    { int s = tid >> 2, db = (tid & 3) * 16;
      const unsigned short* kg = QKV + (long)(b * Td + s0 + s) * 3 * Cd + Cd + h * DHd + db;
      *(u16x8*)&Kt[s][db]     = *(const u16x8*)kg;
      *(u16x8*)&Kt[s][db + 8] = *(const u16x8*)(kg + 8);
      const unsigned short* vg = kg + Cd;
      u16x8 v0 = *(const u16x8*)vg, v1 = *(const u16x8*)(vg + 8);
      #pragma unroll
      for (int i = 0; i < 8; ++i) { Vt[db + i][s] = v0[i]; Vt[db + 8 + i][s] = v1[i]; }
    }
    __syncthreads();

    v8f sc[2]; sc[0] = zero8(); sc[1] = zero8();
    #pragma unroll
    for (int c = 0; c < 2; ++c) {
      v16bf b0 = frag16(&Kt[c * 16 + lrow][half ? 16 : 0]);
      v16bf b1 = frag16(&Kt[c * 16 + lrow][32 + (half ? 16 : 0)]);
      sc[c] = wmma_bf16(aq0, b0, sc[c]);
      sc[c] = wmma_bf16(aq1, b1, sc[c]);
    }
    float mn[8];
    #pragma unroll
    for (int i = 0; i < 8; ++i) mn[i] = mr[i];
    #pragma unroll
    for (int c = 0; c < 2; ++c) {
      #pragma unroll
      for (int i = 0; i < 8; ++i) {
        int tg = t0 + half * 8 + i, sg = s0 + c * 16 + lrow;
        float v = sc[c][i] * SCALE;
        bool blk = (sg > tg) || (dep[(long)(b * Td + tg) * Td + sg] == 0);
        sc[c][i] = blk ? NINF : v;
        mn[i] = fmaxf(mn[i], sc[c][i]);
      }
    }
    #pragma unroll
    for (int i = 0; i < 8; ++i) {
      #pragma unroll
      for (int off = 1; off < 16; off <<= 1) mn[i] = fmaxf(mn[i], __shfl_xor(mn[i], off, 32));
    }
    #pragma unroll
    for (int i = 0; i < 8; ++i) {
      float corr = (mn[i] <= -1e30f) ? 1.f : __expf(mr[i] - mn[i]);
      float p0 = (sc[0][i] <= -1e30f) ? 0.f : __expf(sc[0][i] - mn[i]);
      float p1 = (sc[1][i] <= -1e30f) ? 0.f : __expf(sc[1][i] - mn[i]);
      float ps = p0 + p1;
      #pragma unroll
      for (int off = 1; off < 16; off <<= 1) ps += __shfl_xor(ps, off, 32);
      lr[i] = lr[i] * corr + ps;
      mr[i] = mn[i];
      int r = half * 8 + i;
      Ps[wave][r][lrow]      = f2bf(p0);
      Ps[wave][r][16 + lrow] = f2bf(p1);
      #pragma unroll
      for (int n = 0; n < 4; ++n) o[n][i] *= corr;
    }
    v16bf ap = frag2(&Ps[wave][lrow][ab], &Ps[wave][lrow][ab + 16]);
    #pragma unroll
    for (int n = 0; n < 4; ++n) {
      v16bf bv = frag16(&Vt[n * 16 + lrow][half ? 16 : 0]);
      o[n] = wmma_bf16(ap, bv, o[n]);
    }
    __syncthreads();
  }
  #pragma unroll
  for (int n = 0; n < 4; ++n) {
    #pragma unroll
    for (int i = 0; i < 8; ++i) {
      int r = t0 + half * 8 + i;
      float l = (lr[i] > 0.f) ? lr[i] : 1.f;
      Out[(long)(b * Td + r) * Cd + h * DHd + n * 16 + lrow] = f2bf(o[n][i] / l);
    }
  }
}

// ---------------------------------------------------------------------------
// Hat-stream self attention: merged logits (q_hat·k_star on earlier chunks,
// q_hat·k_hat within chunk), dep mask; values from both streams.
// ---------------------------------------------------------------------------
__global__ __launch_bounds__(128)
void k_attn_hat(const unsigned short* __restrict__ QKVh,
                const unsigned short* __restrict__ QKVs,
                const unsigned char* __restrict__ mstar,
                const unsigned char* __restrict__ mhat,
                const unsigned char* __restrict__ dep,
                unsigned short* __restrict__ Out) {
  __shared__ unsigned short Kts[32][72], Kth[32][72];
  __shared__ unsigned short Vts[64][40], Vth[64][40];
  __shared__ unsigned short Pss[4][16][40], Psh[4][16][40];
  const int tid = threadIdx.x, wave = tid >> 5, lane = tid & 31;
  const int lrow = lane & 15, half = lane >> 4, ab = half ? 16 : 0;
  const int b = blockIdx.x / Hd, h = blockIdx.x % Hd;
  const int t0 = blockIdx.y * 64 + wave * 16;
  const float NINF = -3.0e38f;

  const unsigned short* qrow = QKVh + (long)(b * Td + t0 + lrow) * 3 * Cd + h * DHd;
  v16bf aq0 = frag2(qrow + ab,      qrow + ab + 16);
  v16bf aq1 = frag2(qrow + 32 + ab, qrow + 32 + ab + 16);

  v8f o[4];
  #pragma unroll
  for (int n = 0; n < 4; ++n) o[n] = zero8();
  float mr[8], lr[8];
  #pragma unroll
  for (int i = 0; i < 8; ++i) { mr[i] = NINF; lr[i] = 0.f; }

  for (int s0 = 0; s0 < Td; s0 += 32) {
    { int s = tid >> 2, db = (tid & 3) * 16;
      long rs = (long)(b * Td + s0 + s) * 3 * Cd + Cd + h * DHd + db;
      const unsigned short* kgs = QKVs + rs;
      const unsigned short* kgh = QKVh + rs;
      *(u16x8*)&Kts[s][db]     = *(const u16x8*)kgs;
      *(u16x8*)&Kts[s][db + 8] = *(const u16x8*)(kgs + 8);
      *(u16x8*)&Kth[s][db]     = *(const u16x8*)kgh;
      *(u16x8*)&Kth[s][db + 8] = *(const u16x8*)(kgh + 8);
      u16x8 a0 = *(const u16x8*)(kgs + Cd), a1 = *(const u16x8*)(kgs + Cd + 8);
      u16x8 b0 = *(const u16x8*)(kgh + Cd), b1 = *(const u16x8*)(kgh + Cd + 8);
      #pragma unroll
      for (int i = 0; i < 8; ++i) {
        Vts[db + i][s] = a0[i]; Vts[db + 8 + i][s] = a1[i];
        Vth[db + i][s] = b0[i]; Vth[db + 8 + i][s] = b1[i];
      }
    }
    __syncthreads();

    v8f sS[2], sH[2];
    sS[0] = zero8(); sS[1] = zero8(); sH[0] = zero8(); sH[1] = zero8();
    #pragma unroll
    for (int c = 0; c < 2; ++c) {
      v16bf bs0 = frag16(&Kts[c * 16 + lrow][half ? 16 : 0]);
      v16bf bs1 = frag16(&Kts[c * 16 + lrow][32 + (half ? 16 : 0)]);
      v16bf bh0 = frag16(&Kth[c * 16 + lrow][half ? 16 : 0]);
      v16bf bh1 = frag16(&Kth[c * 16 + lrow][32 + (half ? 16 : 0)]);
      sS[c] = wmma_bf16(aq0, bs0, sS[c]); sS[c] = wmma_bf16(aq1, bs1, sS[c]);
      sH[c] = wmma_bf16(aq0, bh0, sH[c]); sH[c] = wmma_bf16(aq1, bh1, sH[c]);
    }
    unsigned msb[2] = {0u, 0u}, mhb[2] = {0u, 0u};
    float mn[8];
    #pragma unroll
    for (int i = 0; i < 8; ++i) mn[i] = mr[i];
    #pragma unroll
    for (int c = 0; c < 2; ++c) {
      #pragma unroll
      for (int i = 0; i < 8; ++i) {
        int tg = t0 + half * 8 + i, sg = s0 + c * 16 + lrow;
        long mi = (long)(b * Td + tg) * Td + sg;
        bool ms = mstar[mi] != 0, mh = mhat[mi] != 0, dd = dep[mi] != 0;
        float v = mh ? (sH[c][i] * SCALE) : (ms ? (sS[c][i] * SCALE) : NINF);
        if (!dd) v = NINF;
        if (ms) msb[c] |= 1u << i;
        if (mh) mhb[c] |= 1u << i;
        sH[c][i] = v;  // merged logits
        mn[i] = fmaxf(mn[i], v);
      }
    }
    #pragma unroll
    for (int i = 0; i < 8; ++i) {
      #pragma unroll
      for (int off = 1; off < 16; off <<= 1) mn[i] = fmaxf(mn[i], __shfl_xor(mn[i], off, 32));
    }
    #pragma unroll
    for (int i = 0; i < 8; ++i) {
      float corr = (mn[i] <= -1e30f) ? 1.f : __expf(mr[i] - mn[i]);
      float p0 = (sH[0][i] <= -1e30f) ? 0.f : __expf(sH[0][i] - mn[i]);
      float p1 = (sH[1][i] <= -1e30f) ? 0.f : __expf(sH[1][i] - mn[i]);
      float ps = p0 + p1;
      #pragma unroll
      for (int off = 1; off < 16; off <<= 1) ps += __shfl_xor(ps, off, 32);
      lr[i] = lr[i] * corr + ps;
      mr[i] = mn[i];
      int r = half * 8 + i;
      Pss[wave][r][lrow]      = f2bf(((msb[0] >> i) & 1u) ? p0 : 0.f);
      Pss[wave][r][16 + lrow] = f2bf(((msb[1] >> i) & 1u) ? p1 : 0.f);
      Psh[wave][r][lrow]      = f2bf(((mhb[0] >> i) & 1u) ? p0 : 0.f);
      Psh[wave][r][16 + lrow] = f2bf(((mhb[1] >> i) & 1u) ? p1 : 0.f);
      #pragma unroll
      for (int n = 0; n < 4; ++n) o[n][i] *= corr;
    }
    v16bf aps = frag2(&Pss[wave][lrow][ab], &Pss[wave][lrow][ab + 16]);
    v16bf aph = frag2(&Psh[wave][lrow][ab], &Psh[wave][lrow][ab + 16]);
    #pragma unroll
    for (int n = 0; n < 4; ++n) {
      v16bf bvs = frag16(&Vts[n * 16 + lrow][half ? 16 : 0]);
      v16bf bvh = frag16(&Vth[n * 16 + lrow][half ? 16 : 0]);
      o[n] = wmma_bf16(aps, bvs, o[n]);
      o[n] = wmma_bf16(aph, bvh, o[n]);
    }
    __syncthreads();
  }
  #pragma unroll
  for (int n = 0; n < 4; ++n) {
    #pragma unroll
    for (int i = 0; i < 8; ++i) {
      int r = t0 + half * 8 + i;
      float l = (lr[i] > 0.f) ? lr[i] : 1.f;
      Out[(long)(b * Td + r) * Cd + h * DHd + n * 16 + lrow] = f2bf(o[n][i] / l);
    }
  }
}

} // namespace

// ---------------------------------------------------------------------------
// Host orchestration
// ---------------------------------------------------------------------------
extern "C" void kernel_launch(void* const* d_in, const int* in_sizes, int n_in,
                              void* d_out, int out_size, void* d_ws, size_t ws_size,
                              hipStream_t stream) {
  (void)in_sizes; (void)n_in; (void)out_size; (void)ws_size;
  const float* xin[2] = { (const float*)d_in[0], (const float*)d_in[1] };
  const float* cc   = (const float*)d_in[2];
  const unsigned char* m_star = (const unsigned char*)d_in[3];
  const unsigned char* m_hat  = (const unsigned char*)d_in[4];
  const unsigned char* dep    = (const unsigned char*)d_in[5];
  const float* ng  = (const float*)d_in[6];
  const float* nb  = (const float*)d_in[7];
  const float* Wq  = (const float*)d_in[8];  const float* bq   = (const float*)d_in[9];
  const float* Wkv = (const float*)d_in[10]; const float* bkv  = (const float*)d_in[11];
  const float* Wco = (const float*)d_in[12]; const float* bco  = (const float*)d_in[13];
  const float* Wada= (const float*)d_in[14]; const float* bada = (const float*)d_in[15];
  const float* Wqkv= (const float*)d_in[16]; const float* bqkv = (const float*)d_in[17];
  const float* Wo  = (const float*)d_in[18]; const float* bo   = (const float*)d_in[19];
  const float* W1  = (const float*)d_in[20]; const float* b1   = (const float*)d_in[21];
  const float* W2  = (const float*)d_in[22]; const float* b2   = (const float*)d_in[23];

  char* ws = (char*)d_ws;
  size_t off = 0;
  auto alloc = [&](size_t elems, size_t esz) -> void* {
    void* p = ws + off;
    off = (off + elems * esz + 255) & ~(size_t)255;
    return p;
  };
  typedef unsigned short bfp;
  bfp* wq_b   = (bfp*)alloc((size_t)Cd * Cd, 2);
  bfp* wkv_b  = (bfp*)alloc((size_t)Cd * 2 * Cd, 2);
  bfp* wco_b  = (bfp*)alloc((size_t)Cd * Cd, 2);
  bfp* wada_b = (bfp*)alloc((size_t)Cd * 6 * Cd, 2);
  bfp* wqkv_b = (bfp*)alloc((size_t)Cd * 3 * Cd, 2);
  bfp* wo_b   = (bfp*)alloc((size_t)Cd * Cd, 2);
  bfp* w1_b   = (bfp*)alloc((size_t)Cd * DFFd, 2);
  bfp* w2_b   = (bfp*)alloc((size_t)DFFd * Cd, 2);
  bfp* c_b    = (bfp*)alloc((size_t)NCC * Cd, 2);
  bfp* kv_b   = (bfp*)alloc((size_t)NCC * 2 * Cd, 2);
  bfp* ada_b[2]  = { (bfp*)alloc((size_t)NT * 6 * Cd, 2), (bfp*)alloc((size_t)NT * 6 * Cd, 2) };
  bfp* qkv_b[2]  = { (bfp*)alloc((size_t)NT * 3 * Cd, 2), (bfp*)alloc((size_t)NT * 3 * Cd, 2) };
  bfp* yatt_b[2] = { (bfp*)alloc((size_t)NT * Cd, 2),     (bfp*)alloc((size_t)NT * Cd, 2) };
  bfp* ln_b   = (bfp*)alloc((size_t)NT * Cd, 2);
  bfp* q_b    = (bfp*)alloc((size_t)NT * Cd, 2);
  bfp* ca_b   = (bfp*)alloc((size_t)NT * Cd, 2);
  bfp* silu_b = (bfp*)alloc((size_t)NT * Cd, 2);
  bfp* mod_b  = (bfp*)alloc((size_t)NT * Cd, 2);
  bfp* mln_b  = (bfp*)alloc((size_t)NT * Cd, 2);
  bfp* h1_b   = (bfp*)alloc((size_t)NT * DFFd, 2);
  float* projf = (float*)alloc((size_t)NT * Cd, 4);
  float* x1f   = (float*)alloc((size_t)NT * Cd, 4);

  auto cvt = [&](const float* s, bfp* d, int n) {
    k_f32_to_bf16<<<(n + 255) / 256, 256, 0, stream>>>(s, d, n);
  };
  cvt(Wq,   wq_b,   Cd * Cd);
  cvt(Wkv,  wkv_b,  Cd * 2 * Cd);
  cvt(Wco,  wco_b,  Cd * Cd);
  cvt(Wada, wada_b, Cd * 6 * Cd);
  cvt(Wqkv, wqkv_b, Cd * 3 * Cd);
  cvt(Wo,   wo_b,   Cd * Cd);
  cvt(W1,   w1_b,   Cd * DFFd);
  cvt(W2,   w2_b,   DFFd * Cd);
  cvt(cc,   c_b,    NCC * Cd);

  auto gemm = [&](int epi, const bfp* A, const bfp* W, const float* bias, void* out,
                  int M, int N, int K) {
    dim3 g(N / 128, M / 64);
    switch (epi) {
      case 0: k_gemm<0><<<g, 128, 0, stream>>>(A, W, bias, out, M, N, K); break;
      case 1: k_gemm<1><<<g, 128, 0, stream>>>(A, W, bias, out, M, N, K); break;
      case 2: k_gemm<2><<<g, 128, 0, stream>>>(A, W, bias, out, M, N, K); break;
      default:k_gemm<3><<<g, 128, 0, stream>>>(A, W, bias, out, M, N, K); break;
    }
  };

  // kv projection (shared by both streams: depends only on context c)
  gemm(1, c_b, wkv_b, bkv, kv_b, NCC, 2 * Cd, Cd);

  const dim3 agrid(Bd * Hd, Td / 64);

  // Phase 1: per-stream cond-LN -> cross-attn -> AdaLN gates -> mod -> qkv
  for (int s = 0; s < 2; ++s) {
    k_layernorm<0><<<NT, 256, 0, stream>>>(xin[s], ng, nb, nullptr, 0, 0, ln_b);
    gemm(1, ln_b, wq_b, bq, q_b, NT, Cd, Cd);
    k_cross_attn<<<agrid, 128, 0, stream>>>(q_b, kv_b, ca_b);
    gemm(2, ca_b, wco_b, bco, silu_b, NT, Cd, Cd);              // bias + SiLU (AdaLN input)
    gemm(1, silu_b, wada_b, bada, ada_b[s], NT, 6 * Cd, Cd);    // 6C gates (bf16)
    k_layernorm<1><<<NT, 256, 0, stream>>>(xin[s], nullptr, nullptr,
                                           ada_b[s], 0, Cd, mod_b);
    gemm(1, mod_b, wqkv_b, bqkv, qkv_b[s], NT, 3 * Cd, Cd);
  }

  // Phase 2: interleaved self-attention (hat needs star's K/V)
  k_attn_star<<<agrid, 128, 0, stream>>>(qkv_b[0], dep, yatt_b[0]);
  k_attn_hat <<<agrid, 128, 0, stream>>>(qkv_b[1], qkv_b[0], m_star, m_hat, dep, yatt_b[1]);

  // Phase 3: per-stream out-proj + gated residual + MLP
  for (int s = 0; s < 2; ++s) {
    float* outp = (float*)d_out + (size_t)s * NT * Cd;
    gemm(0, yatt_b[s], wo_b, bo, projf, NT, Cd, Cd);
    k_resid_gate<<<(NT * Cd + 255) / 256, 256, 0, stream>>>(xin[s], projf, ada_b[s],
                                                            2 * Cd, x1f, NT * Cd);
    k_layernorm<1><<<NT, 256, 0, stream>>>(x1f, nullptr, nullptr,
                                           ada_b[s], 3 * Cd, 4 * Cd, mln_b);
    gemm(3, mln_b, w1_b, b1, h1_b, NT, DFFd, Cd);               // bias + GELU(tanh)
    gemm(0, h1_b, w2_b, b2, projf, NT, Cd, DFFd);
    k_resid_gate<<<(NT * Cd + 255) / 256, 256, 0, stream>>>(x1f, projf, ada_b[s],
                                                            5 * Cd, outp, NT * Cd);
  }
}
